// TreeTransformerBlock_70308614636051
// MI455X (gfx1250) — matrix operand order repaired and checked by reading, not compiled, not run on hardware
//
#include <hip/hip_runtime.h>
#include <stdint.h>

// ---------------------------------------------------------------------------
// Tree-Transformer block for MI455X (gfx1250, wave32, WMMA 16x16x32 bf16).
// B=4, S=1024, H=1024, NH=16, D=64, FF=4096.
// ---------------------------------------------------------------------------

#define Bsz 4
#define Sq  1024
#define Hd  1024
#define NH  16
#define Dd  64
#define FF  4096
#define NEGV (-1e9f)
#define SQRT1EM9 3.16227766016838e-5f

typedef __attribute__((ext_vector_type(16))) __bf16 bfx16;
typedef __attribute__((ext_vector_type(8)))  __bf16 bfx8;
typedef __attribute__((ext_vector_type(8)))  float  fx8;

__device__ __forceinline__ __bf16 f2bf(float f) {
    unsigned u = __float_as_uint(f);
    unsigned r = u + 0x7FFFu + ((u >> 16) & 1u);   // round-to-nearest-even
    unsigned short h = (unsigned short)(r >> 16);
    __bf16 o; __builtin_memcpy(&o, &h, 2); return o;
}

// Build a 16x32 A-fragment for v_wmma_f32_16x16x32_bf16 from two contiguous
// 8-element groups (ISA 7.12.2: lane half holds K {h*8..h*8+7, h*8+16..h*8+23}).
__device__ __forceinline__ bfx16 ld2(const __bf16* p0, const __bf16* p1) {
    union { bfx16 v; bfx8 h[2]; } u;
    u.h[0] = *(const bfx8*)p0;
    u.h[1] = *(const bfx8*)p1;
    return u.v;
}

__device__ __forceinline__ bfx16 wmma_bf16(bfx16 a, bfx16 b); // unused decl guard

// ---------------------------------------------------------------------------
// Elementwise converts
// ---------------------------------------------------------------------------
__global__ void wconv_kernel(const float* __restrict__ W, __bf16* __restrict__ WT,
                             int K, int N) {
    long i = (long)blockIdx.x * 256 + threadIdx.x;
    if (i >= (long)K * N) return;
    int k = (int)(i / N), n = (int)(i % N);
    WT[(long)n * K + k] = f2bf(W[i]);       // (K,N) row-major -> (N,K) bf16
}

__global__ void f2bf_kernel(const float* __restrict__ X, __bf16* __restrict__ Y, long n) {
    long i = (long)blockIdx.x * 256 + threadIdx.x;
    if (i < n) Y[i] = f2bf(X[i]);
}

// ---------------------------------------------------------------------------
// Fused LayerNorm: y = LN(X (+R)) * g + b ; writes f32 and/or bf16.
// One block per row of H=1024, 256 threads, 4 elems/thread.
// ---------------------------------------------------------------------------
__global__ __launch_bounds__(256)
void ln_kernel(const float* __restrict__ X, const float* __restrict__ R,
               const float* __restrict__ gam, const float* __restrict__ bet,
               float eps, float* __restrict__ outF, __bf16* __restrict__ outB) {
    int row = blockIdx.x, tid = threadIdx.x;
    const float* x = X + (long)row * Hd;
    const float* r = R ? R + (long)row * Hd : nullptr;
    float vals[4], s = 0.f, s2 = 0.f;
    #pragma unroll
    for (int i = 0; i < 4; ++i) {
        int c = tid + i * 256;
        float v = x[c] + (r ? r[c] : 0.f);
        vals[i] = v; s += v; s2 += v * v;
    }
    __shared__ float red[16];
    #pragma unroll
    for (int m = 16; m >= 1; m >>= 1) { s += __shfl_xor(s, m, 32); s2 += __shfl_xor(s2, m, 32); }
    int wid = tid >> 5, lane = tid & 31;
    if (lane == 0) { red[wid] = s; red[8 + wid] = s2; }
    __syncthreads();
    if (tid == 0) {
        float a = 0.f, b = 0.f;
        for (int i = 0; i < 8; ++i) { a += red[i]; b += red[8 + i]; }
        red[0] = a; red[8] = b;
    }
    __syncthreads();
    float mean = red[0] * (1.f / Hd);
    float var  = red[8] * (1.f / Hd) - mean * mean;
    float inv  = rsqrtf(var + eps);
    #pragma unroll
    for (int i = 0; i < 4; ++i) {
        int c = tid + i * 256;
        float y = (vals[i] - mean) * inv * gam[c] + bet[c];
        if (outF) outF[(long)row * Hd + c] = y;
        if (outB) outB[(long)row * Hd + c] = f2bf(y);
    }
}

// ---------------------------------------------------------------------------
// Generic WMMA GEMM: C[M,N] = A_bf16[M,K] @ WT_bf16[N,K]^T + bias
// Block tile 128x64, 256 threads = 8 waves (4x2), wave tile 32x32 (2x2 frags).
// EPI: 0 -> f32 store, 1 -> bf16 store, 2 -> GELU(exact) -> bf16 store.
// ---------------------------------------------------------------------------
template <int EPI>
__global__ __launch_bounds__(256)
void gemm_wmma(const __bf16* __restrict__ A, const __bf16* __restrict__ WT,
               const float* __restrict__ bias,
               float* __restrict__ Cf, __bf16* __restrict__ Cb,
               int M, int N, int K) {
    __shared__ __align__(32) __bf16 As[128][32];
    __shared__ __align__(32) __bf16 Bs[64][32];
    const int tid = threadIdx.x, wid = tid >> 5, lane = tid & 31;
    const int waveM = wid & 3, waveN = wid >> 2;
    const int colL = lane & 15, kh = lane >> 4;
    const int m0 = blockIdx.y * 128, n0 = blockIdx.x * 64;
    fx8 acc[2][2] = {};

    const int ar = tid >> 1, ah = tid & 1;
    for (int k0 = 0; k0 < K; k0 += 32) {
        // Stage tiles (coalesced 32B vector loads)
        *(bfx16*)&As[ar][ah * 16] = *(const bfx16*)&A[(long)(m0 + ar) * K + k0 + ah * 16];
        if (tid < 128)
            *(bfx16*)&Bs[ar][ah * 16] = *(const bfx16*)&WT[(long)(n0 + ar) * K + k0 + ah * 16];
        __builtin_prefetch(&A[(long)(m0 + ar) * K + k0 + 32], 0, 0);
        __syncthreads();

        bfx16 af[2], bfr[2];
        #pragma unroll
        for (int fm = 0; fm < 2; ++fm) {
            const __bf16* p = &As[waveM * 32 + fm * 16 + colL][kh * 8];
            af[fm] = ld2(p, p + 16);
        }
        #pragma unroll
        for (int fn = 0; fn < 2; ++fn)
            bfr[fn] = ld2(&Bs[waveN * 32 + fn * 16 + colL][kh * 8], &Bs[waveN * 32 + fn * 16 + colL][kh * 8 + 16]);

        #pragma unroll
        for (int fm = 0; fm < 2; ++fm)
            #pragma unroll
            for (int fn = 0; fn < 2; ++fn)
                acc[fm][fn] = __builtin_amdgcn_wmma_f32_16x16x32_bf16(
                    false, af[fm], false, bfr[fn], (short)0, acc[fm][fn], false, false);
        __syncthreads();
    }
    // Epilogue: C layout per ISA 7.12.2 (lanes<16: N=lane, M=r; lanes>=16: M=r+8)
    #pragma unroll
    for (int fm = 0; fm < 2; ++fm)
        #pragma unroll
        for (int fn = 0; fn < 2; ++fn) {
            int tm = m0 + waveM * 32 + fm * 16 + kh * 8;
            int tn = n0 + waveN * 32 + fn * 16 + colL;
            float bv = bias ? bias[tn] : 0.f;
            #pragma unroll
            for (int r = 0; r < 8; ++r) {
                float v = acc[fm][fn][r] + bv;
                long idx = (long)(tm + r) * N + tn;
                if (EPI == 0) Cf[idx] = v;
                else if (EPI == 1) Cb[idx] = f2bf(v);
                else { float g = 0.5f * v * (1.f + erff(v * 0.70710678118f)); Cb[idx] = f2bf(g); }
            }
        }
}

// ---------------------------------------------------------------------------
// GroupAttention neighbor softmax: one wave per (b,s) row.
// Only adjacent entries survive the tridiagonal mask, so the softmax row is
// {e_prev, e_next, 1022 x NEG}.
// ---------------------------------------------------------------------------
__global__ __launch_bounds__(256)
void neighbor_kernel(const float* __restrict__ Q, const float* __restrict__ Kk,
                     const float* __restrict__ mask,
                     float* __restrict__ naf, float* __restrict__ nab) {
    int wid = threadIdx.x >> 5, lane = threadIdx.x & 31;
    int row = blockIdx.x * 8 + wid;
    int b = row / Sq, s = row % Sq;
    const float* q  = Q + (long)row * Hd;
    const float* kp = (s > 0)      ? Kk + (long)(row - 1) * Hd : nullptr;
    const float* kn = (s < Sq - 1) ? Kk + (long)(row + 1) * Hd : nullptr;
    float dp = 0.f, dn = 0.f;
    for (int i = lane; i < Hd; i += 32) {
        float qv = q[i];
        if (kp) dp += qv * kp[i];
        if (kn) dn += qv * kn[i];
    }
    #pragma unroll
    for (int m = 16; m >= 1; m >>= 1) { dp += __shfl_xor(dp, m, 32); dn += __shfl_xor(dn, m, 32); }
    if (lane == 0) {
        float pms = mask[(long)b * Sq + s];
        float ep = (s > 0 && pms > 0.5f && mask[(long)b * Sq + s - 1] > 0.5f) ? dp * (1.f / 32.f) : NEGV;
        float en = (s < Sq - 1 && pms > 0.5f && mask[(long)b * Sq + s + 1] > 0.5f) ? dn * (1.f / 32.f) : NEGV;
        float mx = fmaxf(ep, en);
        float Z  = expf(ep - mx) + expf(en - mx) + (float)(Sq - 2) * expf(NEGV - mx);
        nab[row] = expf(ep - mx) / Z;   // na[s, s-1]
        naf[row] = expf(en - mx) / Z;   // na[s, s+1]
    }
}

// tri combine: na_sym = sqrt(na[s,s+1]*na[s+1,s]+1e-9), then prior mix.
__global__ void tri_kernel(const float* __restrict__ naf, const float* __restrict__ nab,
                           const float* __restrict__ prior,
                           float* __restrict__ triU, float* __restrict__ triL) {
    long i = (long)blockIdx.x * 256 + threadIdx.x;
    if (i >= (long)Bsz * (Sq - 1)) return;
    int b = (int)(i / (Sq - 1)), s = (int)(i % (Sq - 1));
    float v = sqrtf(naf[(long)b * Sq + s] * nab[(long)b * Sq + s + 1] + 1e-9f);
    float prU = prior[((long)b * Sq + s) * Sq + s + 1];
    float prL = prior[((long)b * Sq + s + 1) * Sq + s];
    triU[(long)b * Sq + s] = prU + (1.f - prU) * v;
    triL[(long)b * Sq + s] = prL + (1.f - prL) * v;
}

// prefix sum of log(na_tri + 1e-9) per batch (tiny: 4 x 1024 serial).
__global__ void cum_kernel(const float* __restrict__ triU, float* __restrict__ cum) {
    if (threadIdx.x != 0) return;
    int b = blockIdx.x;
    float c = 0.f;
    cum[(long)b * Sq] = 0.f;
    for (int s = 1; s < Sq; ++s) {
        c += logf(triU[(long)b * Sq + s - 1] + 1e-9f);
        cum[(long)b * Sq + s] = c;
    }
}

// neibor_attn output fill (B,S,S)
__global__ void neibor_fill(const float* __restrict__ prior, const float* __restrict__ mask,
                            const float* __restrict__ triU, const float* __restrict__ triL,
                            float* __restrict__ out2) {
    long i = (long)blockIdx.x * 256 + threadIdx.x;
    if (i >= (long)Bsz * Sq * Sq) return;
    int u = (int)(i % Sq);
    long t = i / Sq;
    int s = (int)(t % Sq), b = (int)(t / Sq);
    float v;
    if (u == s + 1)      v = triU[(long)b * Sq + s];
    else if (s == u + 1) v = triL[(long)b * Sq + u];
    else { float pr = prior[i]; v = pr + (1.f - pr) * SQRT1EM9; }
    float pm = (mask[(long)b * Sq + s] > 0.5f && mask[(long)b * Sq + u] > 0.5f) ? 1.f : 0.f;
    out2[i] = v * pm;
}

// group_prob row kernel: g[s,u] = exp(cum[hi]-cum[lo]) + na[s,u] (diag 2+1e-9),
// row-normalized; the triangular matmuls collapse to prefix-product form.
__global__ __launch_bounds__(256)
void gprob_kernel(const float* __restrict__ prior, const float* __restrict__ mask,
                  const float* __restrict__ triU, const float* __restrict__ triL,
                  const float* __restrict__ cum, float* __restrict__ gp) {
    __shared__ float rowbuf[Sq];
    __shared__ float red[8];
    int row = blockIdx.x;
    int s = row % Sq, b = row / Sq;
    int tid = threadIdx.x;
    const float* cumb = cum + (long)b * Sq;
    float partial = 0.f;
    for (int u = tid; u < Sq; u += 256) {
        float gv;
        if (u == s) gv = 2.f + 1e-9f;
        else {
            int lo = min(s, u), hi = max(s, u);
            float gsym = expf(cumb[hi] - cumb[lo]);
            float naval;
            if (u == s + 1)      naval = triU[(long)b * Sq + s];
            else if (s == u + 1) naval = triL[(long)b * Sq + u];
            else { float pr = prior[(long)row * Sq + u]; naval = pr + (1.f - pr) * SQRT1EM9; }
            gv = gsym + naval;
        }
        rowbuf[u] = gv; partial += gv;
    }
    #pragma unroll
    for (int m = 16; m >= 1; m >>= 1) partial += __shfl_xor(partial, m, 32);
    int wid = tid >> 5, lane = tid & 31;
    if (lane == 0) red[wid] = partial;
    __syncthreads();
    float tot = 0.f;
    for (int i = 0; i < 8; ++i) tot += red[i];
    float inv = 1.f / (tot + 1e-9f);
    float pms = mask[(long)b * Sq + s] > 0.5f ? 1.f : 0.f;
    for (int u = tid; u < Sq; u += 256) {
        float pmu = mask[(long)b * Sq + u] > 0.5f ? 1.f : 0.f;
        gp[(long)row * Sq + u] = rowbuf[u] * inv * pms * pmu;
    }
}

// ---------------------------------------------------------------------------
// Strip attention: one block = (b, head, 16-row tile). 128 threads (4 waves).
// Phase 1: scores strip 16x1024 via WMMA QK^T -> LDS (f32).
// Phase 2: row softmax (scaled, masked) x group_prob -> bf16 P strip.
// Phase 3: PV via WMMA, V transposed through LDS per 32-key chunk.
// Dynamic LDS: 64KB scores + 32KB P + 2KB Q + 4KB VT = 104448 B (<320KB/WGP).
// ---------------------------------------------------------------------------
__global__ __launch_bounds__(128)
void attn_kernel(const __bf16* __restrict__ Qb, const __bf16* __restrict__ Kb,
                 const __bf16* __restrict__ Vb, const float* __restrict__ gprob,
                 const float* __restrict__ mask, __bf16* __restrict__ ctx) {
    extern __shared__ char smem[];
    float (*scores)[Sq] = (float (*)[Sq])smem;                         // 16x1024 f32
    __bf16 (*pbuf)[Sq]  = (__bf16 (*)[Sq])(smem + 65536);              // 16x1024 bf16
    __bf16 (*qt)[64]    = (__bf16 (*)[64])(smem + 65536 + 32768);      // 16x64
    __bf16 (*vt)[32]    = (__bf16 (*)[32])(smem + 65536 + 32768 + 2048); // 64x32

    int blk = blockIdx.x;
    int rt = blk & 63, h = (blk >> 6) & 15, b = blk >> 10;
    int s0 = rt * 16;
    int tid = threadIdx.x, wid = tid >> 5, lane = tid & 31;
    int colL = lane & 15, kh = lane >> 4;
    long bs = (long)b * Sq;

    // stage Q tile
    {
        int e = tid * 8, r = e >> 6, d = e & 63;
        *(bfx8*)&qt[r][d] = *(const bfx8*)&Qb[(bs + s0 + r) * Hd + h * 64 + d];
    }
    __syncthreads();

    // Phase 1: scores
    bfx16 a0, a1;
    {
        const __bf16* p = &qt[colL][kh * 8];
        a0 = ld2(p, p + 16);
        a1 = ld2(p + 32, p + 48);
    }
    for (int i = 0; i < 16; ++i) {
        int t0 = (wid * 16 + i) * 16;
        const __bf16* kb = &Kb[(bs + t0 + colL) * Hd + h * 64 + kh * 8];
        bfx16 b0 = ld2(kb, kb + 16);
        bfx16 b1 = ld2(kb + 32, kb + 48);
        fx8 acc = {};
        acc = __builtin_amdgcn_wmma_f32_16x16x32_bf16(false, a0, false, b0, (short)0, acc, false, false);
        acc = __builtin_amdgcn_wmma_f32_16x16x32_bf16(false, a1, false, b1, (short)0, acc, false, false);
        #pragma unroll
        for (int r = 0; r < 8; ++r) scores[kh * 8 + r][t0 + colL] = acc[r];
    }
    __syncthreads();

    // Phase 2: softmax * group_prob
    {
        int row = tid >> 3, sub = tid & 7;
        float mx = -3.4e38f;
        for (int c = sub; c < Sq; c += 8) {
            float sc = scores[row][c] * 0.125f + (1.f - mask[bs + c]) * NEGV;
            scores[row][c] = sc;
            mx = fmaxf(mx, sc);
        }
        #pragma unroll
        for (int m = 4; m >= 1; m >>= 1) mx = fmaxf(mx, __shfl_xor(mx, m, 32));
        float l = 0.f;
        for (int c = sub; c < Sq; c += 8) l += expf(scores[row][c] - mx);
        #pragma unroll
        for (int m = 4; m >= 1; m >>= 1) l += __shfl_xor(l, m, 32);
        float inv = 1.f / l;
        const float* grow = gprob + (bs + s0 + row) * Sq;
        for (int c = sub; c < Sq; c += 8)
            pbuf[row][c] = f2bf(expf(scores[row][c] - mx) * inv * grow[c]);
    }
    __syncthreads();

    // Phase 3: O = P @ V  (each wave owns one 16-wide d tile)
    fx8 oacc = {};
    for (int kc = 0; kc < 32; ++kc) {
        #pragma unroll
        for (int j = 0; j < 2; ++j) {   // stage 32x64 V chunk transposed
            int e = (tid * 2 + j) * 8, kl = e >> 6, d = e & 63;
            bfx8 v8 = *(const bfx8*)&Vb[(bs + kc * 32 + kl) * Hd + h * 64 + d];
            #pragma unroll
            for (int q2 = 0; q2 < 8; ++q2) vt[d + q2][kl] = v8[q2];
        }
        __syncthreads();
        const __bf16* pp = &pbuf[colL][kc * 32 + kh * 8];
        bfx16 af  = ld2(pp, pp + 16);
        bfx16 bfr = ld2(&vt[wid * 16 + colL][kh * 8], &vt[wid * 16 + colL][kh * 8 + 16]);
        oacc = __builtin_amdgcn_wmma_f32_16x16x32_bf16(false, af, false, bfr, (short)0, oacc, false, false);
        __syncthreads();
    }
    #pragma unroll
    for (int r = 0; r < 8; ++r)
        ctx[(bs + s0 + kh * 8 + r) * Hd + h * 64 + wid * 16 + colL] = f2bf(oacc[r]);
}

// ---------------------------------------------------------------------------
// Host-side orchestration
// ---------------------------------------------------------------------------
extern "C" void kernel_launch(void* const* d_in, const int* in_sizes, int n_in,
                              void* d_out, int out_size, void* d_ws, size_t ws_size,
                              hipStream_t stream) {
    const float* x      = (const float*)d_in[0];
    const float* mask   = (const float*)d_in[1];
    const float* prior  = (const float*)d_in[2];
    const float* galng  = (const float*)d_in[3];
    const float* galnb  = (const float*)d_in[4];
    const float* ga_wk  = (const float*)d_in[5];
    const float* ga_bk  = (const float*)d_in[6];
    const float* ga_wq  = (const float*)d_in[7];
    const float* ga_bq  = (const float*)d_in[8];
    const float* wq_w   = (const float*)d_in[9];
    const float* wq_b   = (const float*)d_in[10];
    const float* wk_w   = (const float*)d_in[11];
    const float* wk_b   = (const float*)d_in[12];
    const float* wv_w   = (const float*)d_in[13];
    const float* wv_b   = (const float*)d_in[14];
    const float* wo_w   = (const float*)d_in[15];
    const float* wo_b   = (const float*)d_in[16];
    const float* ln1g   = (const float*)d_in[17];
    const float* ln1b   = (const float*)d_in[18];
    const float* f1_w   = (const float*)d_in[19];
    const float* f1_b   = (const float*)d_in[20];
    const float* f2_w   = (const float*)d_in[21];
    const float* f2_b   = (const float*)d_in[22];
    const float* ln2g   = (const float*)d_in[23];
    const float* ln2b   = (const float*)d_in[24];

    float* out0 = (float*)d_out;                              // (B,S,H)
    float* out1 = (float*)d_out + (long)Bsz * Sq * Hd;        // (B,S,S) neibor_attn

    // --- workspace layout ---
    char* base = (char*)d_ws;
    size_t off = 0;
    auto alloc = [&](size_t bytes) { void* p = base + off; off = (off + bytes + 255) & ~(size_t)255; return p; };
    const long BSH = (long)Bsz * Sq * Hd;                     // 4M
    __bf16* wtGAK = (__bf16*)alloc((long)Hd * Hd * 2);
    __bf16* wtGAQ = (__bf16*)alloc((long)Hd * Hd * 2);
    __bf16* wtQ   = (__bf16*)alloc((long)Hd * Hd * 2);
    __bf16* wtK   = (__bf16*)alloc((long)Hd * Hd * 2);
    __bf16* wtV   = (__bf16*)alloc((long)Hd * Hd * 2);
    __bf16* wtO   = (__bf16*)alloc((long)Hd * Hd * 2);
    __bf16* wtF1  = (__bf16*)alloc((long)Hd * FF * 2);        // (FF,Hd)
    __bf16* wtF2  = (__bf16*)alloc((long)Hd * FF * 2);        // (Hd,FF)
    __bf16* xbf   = (__bf16*)alloc(BSH * 2);
    __bf16* cnbf  = (__bf16*)alloc(BSH * 2);                  // reused as ctx
    float*  qga   = (float*) alloc(BSH * 4);                  // reused as attn
    float*  kga   = (float*) alloc(BSH * 4);                  // reused as h1_f32
    __bf16* qbf   = (__bf16*)alloc(BSH * 2);
    __bf16* kbf   = (__bf16*)alloc(BSH * 2);
    __bf16* vbf   = (__bf16*)alloc(BSH * 2);
    float*  naf   = (float*) alloc((long)Bsz * Sq * 4);
    float*  nab   = (float*) alloc((long)Bsz * Sq * 4);
    float*  triU  = (float*) alloc((long)Bsz * Sq * 4);
    float*  triL  = (float*) alloc((long)Bsz * Sq * 4);
    float*  cum   = (float*) alloc((long)Bsz * Sq * 4);
    float*  gpb   = (float*) alloc((long)Bsz * Sq * Sq * 4);  // reused as ffn_f32
    __bf16* h1bf  = (__bf16*)alloc(BSH * 2);
    __bf16* midbf = (__bf16*)alloc((long)Bsz * Sq * FF * 2);
    __bf16* ctx   = cnbf;                                     // alias (cn dead by then)
    float*  attnf = qga;                                      // alias
    float*  h1f   = kga;                                      // alias
    float*  ffnf  = gpb;                                      // alias

    const int MR = Bsz * Sq;                                  // 4096 rows
    auto gconv = [&](const float* W, __bf16* WT, int K, int N) {
        wconv_kernel<<<(unsigned)(((long)K * N + 255) / 256), 256, 0, stream>>>(W, WT, K, N);
    };

    // 1) weight converts (fp32 -> bf16, transposed to (N,K))
    gconv(ga_wk, wtGAK, Hd, Hd); gconv(ga_wq, wtGAQ, Hd, Hd);
    gconv(wq_w, wtQ, Hd, Hd);    gconv(wk_w, wtK, Hd, Hd);
    gconv(wv_w, wtV, Hd, Hd);    gconv(wo_w, wtO, Hd, Hd);
    gconv(f1_w, wtF1, Hd, FF);   gconv(f2_w, wtF2, FF, Hd);
    f2bf_kernel<<<(unsigned)((BSH + 255) / 256), 256, 0, stream>>>(x, xbf, BSH);

    // 2) GroupAttention LN + q/k projections
    ln_kernel<<<MR, 256, 0, stream>>>(x, nullptr, galng, galnb, 1e-5f, nullptr, cnbf);
    gemm_wmma<0><<<dim3(Hd / 64, MR / 128), 256, 0, stream>>>(cnbf, wtGAK, ga_bk, kga, nullptr, MR, Hd, Hd);
    gemm_wmma<0><<<dim3(Hd / 64, MR / 128), 256, 0, stream>>>(cnbf, wtGAQ, ga_bq, qga, nullptr, MR, Hd, Hd);

    // 3) tridiagonal softmax + prior mix + prefix products + outputs
    neighbor_kernel<<<MR / 8, 256, 0, stream>>>(qga, kga, mask, naf, nab);
    tri_kernel<<<(unsigned)(((long)Bsz * (Sq - 1) + 255) / 256), 256, 0, stream>>>(naf, nab, prior, triU, triL);
    neibor_fill<<<(unsigned)(((long)Bsz * Sq * Sq + 255) / 256), 256, 0, stream>>>(prior, mask, triU, triL, out1);
    cum_kernel<<<Bsz, 32, 0, stream>>>(triU, cum);
    gprob_kernel<<<MR, 256, 0, stream>>>(prior, mask, triU, triL, cum, gpb);

    // 4) Tree-MHA
    gemm_wmma<1><<<dim3(Hd / 64, MR / 128), 256, 0, stream>>>(xbf, wtQ, wq_b, nullptr, qbf, MR, Hd, Hd);
    gemm_wmma<1><<<dim3(Hd / 64, MR / 128), 256, 0, stream>>>(xbf, wtK, wk_b, nullptr, kbf, MR, Hd, Hd);
    gemm_wmma<1><<<dim3(Hd / 64, MR / 128), 256, 0, stream>>>(xbf, wtV, wv_b, nullptr, vbf, MR, Hd, Hd);
    {
        unsigned grid = Bsz * NH * (Sq / 16);                  // 4096 blocks
        size_t lds = 65536 + 32768 + 2048 + 4096;              // 104448 B
        attn_kernel<<<grid, 128, lds, stream>>>(qbf, kbf, vbf, gpb, mask, ctx);
    }
    gemm_wmma<0><<<dim3(Hd / 64, MR / 128), 256, 0, stream>>>(ctx, wtO, wo_b, attnf, nullptr, MR, Hd, Hd);

    // 5) post-LN + FFN + final LN
    ln_kernel<<<MR, 256, 0, stream>>>(x, attnf, ln1g, ln1b, 1e-12f, h1f, h1bf);
    gemm_wmma<2><<<dim3(FF / 64, MR / 128), 256, 0, stream>>>(h1bf, wtF1, f1_b, nullptr, midbf, MR, FF, Hd);
    gemm_wmma<0><<<dim3(Hd / 64, MR / 128), 256, 0, stream>>>(midbf, wtF2, f2_b, ffnf, nullptr, MR, Hd, FF);
    ln_kernel<<<MR, 256, 0, stream>>>(h1f, ffnf, ln2g, ln2b, 1e-12f, out0, nullptr);
}
